// SLSTMCell_3642132267141
// MI455X (gfx1250) — compile-verified, matrix operand order checked by reading, not run on hardware
//
#include <hip/hip_runtime.h>
#include <stddef.h>

// ---------------- types ----------------
typedef __bf16 bf16;
typedef __attribute__((ext_vector_type(16))) __bf16 v16bf;
typedef __attribute__((ext_vector_type(8)))  __bf16 v8bf;
typedef __attribute__((ext_vector_type(8)))  float  v8f;
typedef int vi4 __attribute__((vector_size(16)));   // matches builtin's V4i pointee

#define B_  64
#define L_  512
#define H_  256
#define NL  20
#define BL  (B_ * L_)              // 32768
#define BLH ((size_t)BL * H_)      // 8388608
#define KBIG 1280
#define NBIG 1536                  // 6 gates (u gate is dead code in reference)
#define KFGI 512
#define NFGI 256
#define KA   1280                  // sA feature width
#define KPAD 48                    // padded k-extent of LDS tiles (32 + 16 pad)

__device__ __forceinline__ bf16 f2bf(float f) {
    union { float f; unsigned u; } v; v.f = f;
    unsigned r = v.u + 0x7FFFu + ((v.u >> 16) & 1u);   // RNE
    unsigned short h = (unsigned short)(r >> 16);
    bf16 b; __builtin_memcpy(&b, &h, 2); return b;
}
__device__ __forceinline__ float bf2f(bf16 x) {
    unsigned short s; __builtin_memcpy(&s, &x, 2);
    union { unsigned u; float f; } v; v.u = ((unsigned)s) << 16; return v.f;
}
__device__ __forceinline__ float sigm(float x) { return 1.0f / (1.0f + __expf(-x)); }

// ---------------- async LDS copy (CDNA5 GLOBAL_LOAD_ASYNC_TO_LDS_B128) ----------------
#if __has_builtin(__builtin_amdgcn_global_load_async_to_lds_b128) && \
    __has_builtin(__builtin_amdgcn_s_wait_asynccnt)
#define USE_ASYNC_CP 1
#else
#define USE_ASYNC_CP 0
#endif

__device__ __forceinline__ void cp16_lds(void* lds_dst, const void* gsrc) {
#if USE_ASYNC_CP
    typedef __attribute__((address_space(1))) vi4 gvi4;
    typedef __attribute__((address_space(3))) vi4 lvi4;
    __builtin_amdgcn_global_load_async_to_lds_b128(
        (gvi4*)(uintptr_t)gsrc,
        (lvi4*)(uintptr_t)(unsigned)(uintptr_t)lds_dst,
        0, 0);
#else
    *(uint4*)lds_dst = *(const uint4*)gsrc;
#endif
}
__device__ __forceinline__ void cp_commit() {
#if USE_ASYNC_CP
    __builtin_amdgcn_s_wait_asynccnt(0);
#endif
}

// ---------------- weight packing (pre-swizzled for contiguous LDS tiles) ----------------
// Wpack[((k>>5)*N + n)*KPAD + (k&31)] = bf16(W[k][n]);  tile (ktile,N0) is one
// contiguous 128*KPAD*2-byte block -> direct async copy into LDS.
__global__ __launch_bounds__(256) void pack_w7(const float* __restrict__ Wi, const float* __restrict__ Wl,
                        const float* __restrict__ Wr, const float* __restrict__ Wf,
                        const float* __restrict__ Ws, const float* __restrict__ Wo,
                        bf16* __restrict__ Wp) {
    int idx = blockIdx.x * 256 + threadIdx.x;
    const int total = KBIG * NBIG;
    if (idx >= total) return;
    int k = idx / NBIG, n = idx - k * NBIG;
    int g = n >> 8, c = n & 255;
    const float* W[6] = {Wi, Wl, Wr, Wf, Ws, Wo};
    Wp[((size_t)(k >> 5) * NBIG + n) * KPAD + (k & 31)] = f2bf(W[g][k * 256 + c]);
}

__global__ __launch_bounds__(256) void pack_wfi(const float* __restrict__ Wfi, bf16* __restrict__ Wp) {
    int idx = blockIdx.x * 256 + threadIdx.x;
    const int total = KFGI * NFGI;
    if (idx >= total) return;
    int k = idx / NFGI, n = idx - k * NFGI;
    Wp[((size_t)(k >> 5) * NFGI + n) * KPAD + (k & 31)] = f2bf(Wfi[k * NFGI + n]);
}

__global__ __launch_bounds__(256) void pack_b7(const float* __restrict__ bi, const float* __restrict__ bl,
                        const float* __restrict__ br, const float* __restrict__ bf,
                        const float* __restrict__ bs, const float* __restrict__ bo,
                        float* __restrict__ b7) {
    int idx = blockIdx.x * 256 + threadIdx.x;
    if (idx >= NBIG) return;
    const float* Bv[6] = {bi, bl, br, bf, bs, bo};
    b7[idx] = Bv[idx >> 8][idx & 255];
}

// ---------------- embedding + init ----------------
// sA columns: [0,256)=h_before [256,512)=h [512,768)=h_after [768,1024)=we [1024,1280)=hg_rep
__global__ __launch_bounds__(256) void embed_init(const int* __restrict__ ids, const float* __restrict__ masks,
                           const float* __restrict__ emb,
                           bf16* __restrict__ sA, float* __restrict__ h, float* __restrict__ c) {
    size_t idx = (size_t)blockIdx.x * 256 + threadIdx.x;
    if (idx >= BLH) return;
    int ch = (int)(idx & 255);
    size_t bl = idx >> 8;
    float m = masks[bl];
    float e = emb[(size_t)ids[bl] * H_ + ch];
    float w = e * m;
    sA[bl * KA + 768 + ch] = f2bf(w);       // we (layer-invariant)
    h[idx] = w * m; c[idx] = w * m;
}

// broadcast current hg into sA hg_rep columns (per layer, before consumers)
__global__ __launch_bounds__(256) void set_hg_sA(const float* __restrict__ hg, bf16* __restrict__ sA) {
    size_t idx = (size_t)blockIdx.x * 256 + threadIdx.x;
    if (idx >= BLH) return;
    int ch = (int)(idx & 255);
    size_t bl = idx >> 8;
    int b = (int)(bl >> 9);
    sA[bl * KA + 1024 + ch] = f2bf(hg[b * H_ + ch]);
}

// mean over L: out[b,ch] = mean_l in[b,l,ch]; optional second copy
__global__ __launch_bounds__(256) void mean_L(const float* __restrict__ in, float* __restrict__ out,
                       float* __restrict__ out2) {
    int b = blockIdx.x, ch = threadIdx.x;
    const float* p = in + (size_t)b * L_ * H_ + ch;
    float s = 0.f;
    for (int l = 0; l < L_; ++l) s += p[(size_t)l * H_];
    s *= (1.0f / (float)L_);
    out[b * H_ + ch] = s;
    if (out2) out2[b * H_ + ch] = s;
}

// f_g = sigmoid([hg,avg_h] @ W_fg + b), o_g likewise
__global__ __launch_bounds__(256) void gate_global(const float* __restrict__ hg, const float* __restrict__ avg_h,
                            const float* __restrict__ Wfg, const float* __restrict__ bfg,
                            const float* __restrict__ Wog, const float* __restrict__ bog,
                            float* __restrict__ f_g, float* __restrict__ o_g) {
    int idx = blockIdx.x * 256 + threadIdx.x;   // B_*H_
    int b = idx >> 8, o = idx & 255;
    float af = bfg[o], ao = bog[o];
    const float* hgb = hg + b * H_;
    const float* avb = avg_h + b * H_;
    for (int k = 0; k < H_; ++k) { float g = hgb[k]; af += g * Wfg[k * H_ + o]; ao += g * Wog[k * H_ + o]; }
    for (int k = 0; k < H_; ++k) { float g = avb[k]; af += g * Wfg[(H_ + k) * H_ + o]; ao += g * Wog[(H_ + k) * H_ + o]; }
    f_g[idx] = sigm(af); o_g[idx] = sigm(ao);
}

// cumsum over L: fills sA h_before/h/h_after (bf16) and ccs/cc_last (f32)
__global__ __launch_bounds__(256) void scan_L(const float* __restrict__ h, const float* __restrict__ c,
                       bf16* __restrict__ sA, float* __restrict__ ccs, float* __restrict__ cc_last) {
    int b = blockIdx.x, ch = threadIdx.x;
    size_t base = (size_t)b * L_ * H_ + ch;
    float th = 0.f;
    for (int l = 0; l < L_; ++l) th += h[base + (size_t)l * H_];
    float rh = 0.f, rc = 0.f;
    for (int l = 0; l < L_; ++l) {
        size_t o = base + (size_t)l * H_;
        float hv = h[o], cv = c[o];
        size_t ar = ((size_t)(b * L_ + l)) * KA + ch;
        sA[ar]       = f2bf(rh);        // h_before (exclusive scan)
        rh += hv;
        sA[ar + 256] = f2bf(hv);        // h
        sA[ar + 512] = f2bf(th - rh);   // h_after
        rc += cv;
        ccs[o] = rc;
    }
    cc_last[b * H_ + ch] = rc;
}

// ---------------- WMMA GEMM (bf16, f32 acc), async double-buffered tiles ----------------
// BM=128 BN=128 BK=32, 8 waves (2 M x 4 N), each wave 64x32 -> 4x2 wmma tiles.
#define MODE_FGI 0
#define MODE_BIG 1

template <int MODE>
__global__ __launch_bounds__(256)
void wmma_gemm(const bf16* __restrict__ sA,   // BL x KA bf16 (feature matrix)
               const bf16* __restrict__ Wp,   // packed (K/32, N, KPAD) bf16
               const float* __restrict__ bias,
               float* __restrict__ out_f32,   // MODE_FGI result (sigmoid applied)
               bf16*  __restrict__ out_bf,    // MODE_BIG result (sigmoid applied)
               int K, int N) {
    const int M0 = blockIdx.x * 128;
    const int N0 = blockIdx.y * 128;
    const int tid  = threadIdx.x;
    const int lane = tid & 31;
    const int wave = tid >> 5;
    const int waveM = wave >> 2;     // 0..1
    const int waveN = wave & 3;      // 0..3
    const int row  = lane & 15;
    const int half = lane >> 4;

    __shared__ __align__(16) bf16 As[2][128 * KPAD];
    __shared__ __align__(16) bf16 Bs[2][128 * KPAD];

    // A tile: 128 rows x 32 k = 512 x 16B chunks (2 per thread), contiguous per row-part
    auto issueA = [&](int kt, int buf) {
        int cb = (MODE == MODE_FGI && kt < 256) ? (kt + 1024) : kt;   // column remap
        #pragma unroll
        for (int i = 0; i < 2; ++i) {
            int chn = tid + i * 256;
            int r = chn >> 2, part = chn & 3;
            const char* src = (const char*)sA + ((size_t)(M0 + r) * KA + cb) * 2 + part * 16;
            char* dst = (char*)&As[buf][0] + r * (KPAD * 2) + part * 16;
            cp16_lds(dst, src);
        }
    };
    // B tile: one contiguous 128*KPAD*2 = 12288B block = 768 x 16B chunks (3 per thread)
    auto issueB = [&](int kt, int buf) {
        const char* base = (const char*)Wp + ((size_t)(kt >> 5) * N + N0) * (KPAD * 2);
        #pragma unroll
        for (int i = 0; i < 3; ++i) {
            int chn = tid + i * 256;
            cp16_lds((char*)&Bs[buf][0] + chn * 16, base + chn * 16);
        }
    };

    v8f acc[4][2];
    const v8f vzero = {0.f, 0.f, 0.f, 0.f, 0.f, 0.f, 0.f, 0.f};
    #pragma unroll
    for (int i = 0; i < 4; ++i)
        #pragma unroll
        for (int j = 0; j < 2; ++j) acc[i][j] = vzero;

    issueB(0, 0);
    issueA(0, 0);
    cp_commit();
    __syncthreads();

    const int nk = K / 32;
    for (int it = 0; it < nk; ++it) {
        const int cur = it & 1, nxt = cur ^ 1;
        if (it + 1 < nk) {                 // prefetch next tiles (DMA overlaps WMMAs)
            issueB((it + 1) * 32, nxt);
            issueA((it + 1) * 32, nxt);
        }
        // ---- compute: 8 WMMAs per wave ----
        v16bf bfrag[2];
        #pragma unroll
        for (int wn = 0; wn < 2; ++wn) {
            const bf16* bp = &Bs[cur][(waveN * 32 + wn * 16 + row) * KPAD];
            v8bf lo = *(const v8bf*)(bp + half * 8);
            v8bf hi = *(const v8bf*)(bp + 16 + half * 8);
            #pragma unroll
            for (int e = 0; e < 8; ++e) { bfrag[wn][e] = lo[e]; bfrag[wn][8 + e] = hi[e]; }
        }
        #pragma unroll
        for (int wm = 0; wm < 4; ++wm) {
            const bf16* ap = &As[cur][(waveM * 64 + wm * 16 + row) * KPAD];
            v8bf lo = *(const v8bf*)(ap + half * 8);
            v8bf hi = *(const v8bf*)(ap + 16 + half * 8);
            v16bf afrag;
            #pragma unroll
            for (int e = 0; e < 8; ++e) { afrag[e] = lo[e]; afrag[8 + e] = hi[e]; }
            #pragma unroll
            for (int wn = 0; wn < 2; ++wn) {
                acc[wm][wn] = __builtin_amdgcn_wmma_f32_16x16x32_bf16(
                    false, afrag, false, bfrag[wn], (short)0, acc[wm][wn], false, false);
            }
        }
        if (it + 1 < nk) cp_commit();      // wave's async copies done before barrier
        __syncthreads();
    }

    // ---- epilogue: bias + sigmoid ----
    #pragma unroll
    for (int wm = 0; wm < 4; ++wm) {
        #pragma unroll
        for (int wn = 0; wn < 2; ++wn) {
            #pragma unroll
            for (int r = 0; r < 8; ++r) {
                int gm = M0 + waveM * 64 + wm * 16 + half * 8 + r;
                int gn = N0 + waveN * 32 + wn * 16 + row;
                float v = sigm(acc[wm][wn][r] + bias[gn]);
                if (MODE == MODE_FGI) out_f32[(size_t)gm * N + gn] = v;
                else                  out_bf [(size_t)gm * N + gn] = f2bf(v);
            }
        }
    }
}

// ---------------- global cell: softmax over (L f_g_i entries + f_g) ----------------
__global__ __launch_bounds__(256) void global_cell(const float* __restrict__ fgi, const float* __restrict__ f_g,
                            const float* __restrict__ masks, const float* __restrict__ c,
                            const float* __restrict__ cg, float* __restrict__ c_g_out) {
    int b = blockIdx.x, ch = threadIdx.x;
    const float* fp = fgi + (size_t)b * L_ * H_ + ch;
    const float* mp = masks + b * L_;
    float fgv = f_g[b * H_ + ch];
    float mx = fgv;
    for (int l = 0; l < L_; ++l) {
        float mss = mp[l] * 1e25f - 1e25f;
        mx = fmaxf(mx, fp[(size_t)l * H_] + mss);
    }
    float sum = __expf(fgv - mx);
    for (int l = 0; l < L_; ++l) {
        float mss = mp[l] * 1e25f - 1e25f;
        sum += __expf(fp[(size_t)l * H_] + mss - mx);
    }
    float inv = 1.0f / sum;
    float acc = __expf(fgv - mx) * inv * cg[b * H_ + ch];
    const float* cp = c + (size_t)b * L_ * H_ + ch;
    for (int l = 0; l < L_; ++l) {
        float mss = mp[l] * 1e25f - 1e25f;
        acc += __expf(fp[(size_t)l * H_] + mss - mx) * inv * cp[(size_t)l * H_];
    }
    c_g_out[b * H_ + ch] = acc;
}

// ---------------- joint 5-gate x L softmax + h/c update ----------------
__global__ __launch_bounds__(256) void gate_update(const bf16* __restrict__ g, const float* __restrict__ masks,
                            const float* __restrict__ ccs, const float* __restrict__ cc_last,
                            float* __restrict__ h, float* __restrict__ c) {
    int b = blockIdx.x, ch = threadIdx.x;
    const bf16* gp = g + (size_t)b * L_ * NBIG + ch;
    const float* mp = masks + b * L_;
    float mx = -1e30f;
    for (int l = 0; l < L_; ++l) {
        float mss = mp[l] * 1e25f - 1e25f;
        const bf16* q = gp + (size_t)l * NBIG;
        #pragma unroll
        for (int G = 0; G < 5; ++G) mx = fmaxf(mx, bf2f(q[G * 256]) + mss);
    }
    float sum = 0.f;
    for (int l = 0; l < L_; ++l) {
        float mss = mp[l] * 1e25f - 1e25f;
        const bf16* q = gp + (size_t)l * NBIG;
        #pragma unroll
        for (int G = 0; G < 5; ++G) sum += __expf(bf2f(q[G * 256]) + mss - mx);
    }
    float inv = 1.0f / sum;
    size_t base = (size_t)b * L_ * H_ + ch;
    float ccl = cc_last[b * H_ + ch];
    for (int l = 0; l < L_; ++l) {
        float mss = mp[l] * 1e25f - 1e25f;
        const bf16* q = gp + (size_t)l * NBIG;
        float gl = __expf(bf2f(q[1 * 256]) + mss - mx) * inv;
        float gr = __expf(bf2f(q[2 * 256]) + mss - mx) * inv;
        float go = bf2f(q[5 * 256]);
        size_t o = base + (size_t)l * H_;
        float ccv = ccs[o];
        float cb = ccv - c[o];
        float ca = ccl - ccv;
        float cn = gl * cb + gr * ca;
        c[o] = cn;
        h[o] = go * tanhf(cn);
    }
}

__global__ __launch_bounds__(256) void update_global(const float* __restrict__ o_g, const float* __restrict__ c_g,
                              float* __restrict__ hg, float* __restrict__ cg) {
    int idx = blockIdx.x * 256 + threadIdx.x;
    if (idx >= B_ * H_) return;
    float v = c_g[idx];
    hg[idx] = o_g[idx] * tanhf(v);
    cg[idx] = v;
}

// ---------------- logits: (h + hg) @ W_c + b_c ----------------
__global__ __launch_bounds__(256) void logits_k(const float* __restrict__ h, const float* __restrict__ hg,
                         const float* __restrict__ Wc, const float* __restrict__ bc,
                         float* __restrict__ out) {
    int idx = blockIdx.x * 256 + threadIdx.x;   // BL * NL
    if (idx >= BL * NL) return;
    int rowi = idx / NL, n = idx - rowi * NL;
    int b = rowi >> 9;
    const float* hp  = h  + (size_t)rowi * H_;
    const float* hgp = hg + b * H_;
    float a = bc[n];
    for (int k = 0; k < H_; ++k) a += (hp[k] + hgp[k]) * Wc[k * NL + n];
    out[idx] = a;
}

// ---------------- host ----------------
extern "C" void kernel_launch(void* const* d_in, const int* in_sizes, int n_in,
                              void* d_out, int out_size, void* d_ws, size_t ws_size,
                              hipStream_t stream) {
    (void)in_sizes; (void)n_in; (void)out_size; (void)ws_size;
    const int*   ids   = (const int*)  d_in[0];
    const float* masks = (const float*)d_in[1];
    const float* emb   = (const float*)d_in[2];
    const float* W_i = (const float*)d_in[3],  *b_i = (const float*)d_in[4];
    const float* W_l = (const float*)d_in[5],  *b_l = (const float*)d_in[6];
    const float* W_r = (const float*)d_in[7],  *b_r = (const float*)d_in[8];
    const float* W_f = (const float*)d_in[9],  *b_f = (const float*)d_in[10];
    const float* W_s = (const float*)d_in[11], *b_s = (const float*)d_in[12];
    const float* W_o = (const float*)d_in[13], *b_o = (const float*)d_in[14];
    // d_in[15,16]: W_u, b_u -> provably unused by the reference output
    const float* W_fg = (const float*)d_in[17], *b_fg = (const float*)d_in[18];
    const float* W_fi = (const float*)d_in[19], *b_fi = (const float*)d_in[20];
    const float* W_og = (const float*)d_in[21], *b_og = (const float*)d_in[22];
    const float* W_c  = (const float*)d_in[23], *b_c  = (const float*)d_in[24];
    float* out = (float*)d_out;

    // workspace carve
    size_t off = 0;
    char* base = (char*)d_ws;
    auto carve = [&](size_t bytes) -> void* {
        void* p = base + off;
        off += (bytes + 255) & ~(size_t)255;
        return p;
    };
    float* h    = (float*)carve(BLH * 4);
    float* c    = (float*)carve(BLH * 4);
    float* ccs  = (float*)carve(BLH * 4);
    float* fgi  = (float*)carve(BLH * 4);
    bf16*  sA   = (bf16*) carve((size_t)BL * KA * 2);            // 84 MB feature matrix
    bf16*  g    = (bf16*) carve((size_t)BL * NBIG * 2);          // 100 MB gates
    bf16*  Wp7  = (bf16*) carve((size_t)(KBIG / 32) * NBIG * KPAD * 2);
    bf16*  Wpfi = (bf16*) carve((size_t)(KFGI / 32) * NFGI * KPAD * 2);
    float* b7   = (float*)carve(NBIG * 4);
    float* hg      = (float*)carve(B_ * H_ * 4);
    float* cg      = (float*)carve(B_ * H_ * 4);
    float* avg_h   = (float*)carve(B_ * H_ * 4);
    float* f_g     = (float*)carve(B_ * H_ * 4);
    float* o_g     = (float*)carve(B_ * H_ * 4);
    float* c_g     = (float*)carve(B_ * H_ * 4);
    float* cc_last = (float*)carve(B_ * H_ * 4);

    // weights -> bf16, pre-swizzled for contiguous LDS tiles
    pack_w7<<<(KBIG * NBIG + 255) / 256, 256, 0, stream>>>(W_i, W_l, W_r, W_f, W_s, W_o, Wp7);
    pack_b7<<<(NBIG + 255) / 256, 256, 0, stream>>>(b_i, b_l, b_r, b_f, b_s, b_o, b7);
    pack_wfi<<<(KFGI * NFGI + 255) / 256, 256, 0, stream>>>(W_fi, Wpfi);

    // init: embedding -> we column of sA, h=c=we0*m*m, hg=cg=mean_L(h)
    embed_init<<<(unsigned)((BLH + 255) / 256), 256, 0, stream>>>(ids, masks, emb, sA, h, c);
    mean_L<<<B_, H_, 0, stream>>>(h, hg, cg);

    for (int layer = 0; layer < 3; ++layer) {
        // broadcast current hg into sA; build h_before/h/h_after + c-scan
        set_hg_sA<<<(unsigned)((BLH + 255) / 256), 256, 0, stream>>>(hg, sA);
        scan_L<<<B_, H_, 0, stream>>>(h, c, sA, ccs, cc_last);
        // avg_h, then f_g / o_g from ga = [hg, avg_h]
        mean_L<<<B_, H_, 0, stream>>>(h, avg_h, nullptr);
        gate_global<<<(B_ * H_) / 256, 256, 0, stream>>>(hg, avg_h, W_fg, b_fg, W_og, b_og, f_g, o_g);
        // f_g_i = sigmoid([hg_rep, h] @ W_fi + b_fi)   (WMMA, K=512, N=256)
        {
            dim3 grid(BL / 128, NFGI / 128);
            wmma_gemm<MODE_FGI><<<grid, 256, 0, stream>>>(sA, Wpfi, b_fi, fgi, nullptr, KFGI, NFGI);
        }
        // c_g from softmax over [f_g_i + mss ; f_g]
        global_cell<<<B_, H_, 0, stream>>>(fgi, f_g, masks, c, cg, c_g);
        // g = sigmoid(s_in @ W7 + b7)   (WMMA, K=1280, N=1536)
        {
            dim3 grid(BL / 128, NBIG / 128);
            wmma_gemm<MODE_BIG><<<grid, 256, 0, stream>>>(sA, Wp7, b7, nullptr, g, KBIG, NBIG);
        }
        // joint softmax over 5 gates x L, then c/h update
        gate_update<<<B_, H_, 0, stream>>>(g, masks, ccs, cc_last, h, c);
        // hg = o_g * tanh(c_g); cg = c_g   (after all hg_rep consumers)
        update_global<<<(B_ * H_) / 256, 256, 0, stream>>>(o_g, c_g, hg, cg);
    }

    // logits = (h + hg) @ W_c + b_c
    logits_k<<<(BL * NL + 255) / 256, 256, 0, stream>>>(h, hg, W_c, b_c, out);
}
